// EdgeAwareFeatureEnhancer_81157702025983
// MI455X (gfx1250) — matrix operand order repaired and checked by reading, not compiled, not run on hardware
//
#include <hip/hip_runtime.h>
#include <math.h>

// ---------------------------------------------------------------------------
// EdgeAwareFeatureEnhancer for MI455X (gfx1250, wave32, WMMA)
// B=16, C=256, Ch=128, H=W=128, fp32 in/out.
// Memory-bound (~1 GB total traffic @ 23.3 TB/s); GEMMs on v_wmma f16.
// ---------------------------------------------------------------------------

#define BATCH 16
#define CIN   256
#define CH    128
#define HH    128
#define WW    128
#define EPSV  1e-5f

typedef __attribute__((ext_vector_type(16))) _Float16    v16h;
typedef __attribute__((ext_vector_type(8)))  float       v8f;
typedef __attribute__((ext_vector_type(4)))  unsigned    u32x4;

// Workspace layout (bytes):
//   edge_t f16 [B][H][C][W]                  : 134,217,728
//   W1f    f16 [128][256] (BN-folded)        :      65,536
//   W2f    f16 [256][128] (BN-folded)        :      65,536
//   bias1f f32 [128]                          :         512
//   bias2f f32 [256]                          :       1,024
#define WS_EDGE   ((size_t)0)
#define WS_W1F    ((size_t)134217728)
#define WS_W2F    (WS_W1F + 65536)
#define WS_B1F    (WS_W2F + 65536)
#define WS_B2F    (WS_B1F + 512)

// ---------------------------------------------------------------------------
// Pass 0: fold BN into weights/biases, convert weights to f16.
// h = relu((s1*W1)*edge + (b1*s1 + bb1 - m1*s1)) etc.
// ---------------------------------------------------------------------------
__global__ void fold_kernel(const float* __restrict__ w1, const float* __restrict__ b1,
                            const float* __restrict__ g1, const float* __restrict__ bt1,
                            const float* __restrict__ m1, const float* __restrict__ v1,
                            const float* __restrict__ w2, const float* __restrict__ b2,
                            const float* __restrict__ g2, const float* __restrict__ bt2,
                            const float* __restrict__ m2, const float* __restrict__ v2,
                            _Float16* __restrict__ W1f, _Float16* __restrict__ W2f,
                            float* __restrict__ bias1f, float* __restrict__ bias2f) {
  int i = blockIdx.x * 256 + threadIdx.x;        // 65536 threads total
  if (i < CH * CIN) {                            // W1: [128][256]
    int o = i >> 8;
    float s = g1[o] * rsqrtf(v1[o] + EPSV);
    W1f[i] = (_Float16)(w1[i] * s);
  } else {                                       // W2: [256][128]
    int j = i - CH * CIN;
    int o = j >> 7;
    float s = g2[o] * rsqrtf(v2[o] + EPSV);
    W2f[j] = (_Float16)(w2[j] * s);
  }
  if (i < CH) {
    float s = g1[i] * rsqrtf(v1[i] + EPSV);
    bias1f[i] = b1[i] * s + (bt1[i] - m1[i] * s);
  }
  if (i < CIN) {
    float s = g2[i] * rsqrtf(v2[i] + EPSV);
    bias2f[i] = b2[i] * s + (bt2[i] - m2[i] * s);
  }
}

// ---------------------------------------------------------------------------
// Pass 1: edge = x - boxavg11(x), per (b,c) plane, separable sliding window.
// Output f16, transposed to [b][h][c][w] so pass 2 reads contiguous rows.
// LDS rows padded to 129 floats -> conflict-free column/row sweeps.
// ---------------------------------------------------------------------------
#define P1S 129
__global__ __launch_bounds__(256) void edge_kernel(const float* __restrict__ x,
                                                   _Float16* __restrict__ edge_t) {
  extern __shared__ float sm1[];
  float* xs = sm1;                 // [128][129]
  float* vs = sm1 + HH * P1S;      // [128][129]

  const int bc = blockIdx.x;
  const int b  = bc >> 8;
  const int c  = bc & 255;
  const float* xp = x + (size_t)bc * (HH * WW);

  for (int i = threadIdx.x; i < HH * WW; i += 256) {
    int h = i >> 7, w = i & 127;
    xs[h * P1S + w] = xp[i];
  }
  __syncthreads();

  // Vertical 11-tap sliding sum: thread -> (column w, half of h)
  {
    const int w  = threadIdx.x & 127;
    const int h0 = (threadIdx.x >> 7) * 64;
    float s = 0.f;
    for (int hh = h0 - 5; hh <= h0 + 5; ++hh)
      if (hh >= 0 && hh < HH) s += xs[hh * P1S + w];
    vs[h0 * P1S + w] = s;
    for (int h = h0 + 1; h < h0 + 64; ++h) {
      int ha = h + 5, hr = h - 6;
      if (ha < HH) s += xs[ha * P1S + w];
      if (hr >= 0) s -= xs[hr * P1S + w];
      vs[h * P1S + w] = s;
    }
  }
  __syncthreads();

  // Horizontal 11-tap sliding sum + edge, in place into xs
  {
    const int h  = threadIdx.x >> 1;
    const int w0 = (threadIdx.x & 1) * 64;
    const float inv = 1.0f / 121.0f;
    float s = 0.f;
    for (int ww = w0 - 5; ww <= w0 + 5; ++ww)
      if (ww >= 0 && ww < WW) s += vs[h * P1S + ww];
    xs[h * P1S + w0] -= s * inv;
    for (int w = w0 + 1; w < w0 + 64; ++w) {
      int wa = w + 5, wr = w - 6;
      if (wa < WW) s += vs[h * P1S + wa];
      if (wr >= 0) s -= vs[h * P1S + wr];
      xs[h * P1S + w] -= s * inv;
    }
  }
  __syncthreads();

  // Coalesced f16 store, transposed layout [b][h][c][w]
  for (int i = threadIdx.x; i < HH * WW; i += 256) {
    int h = i >> 7, w = i & 127;
    edge_t[(((size_t)b * HH + h) * CIN + c) * WW + w] = (_Float16)xs[h * P1S + w];
  }
}

// ---------------------------------------------------------------------------
// Pass 2: per (b,h) row: edge -> conv1(relu) -> conv2(sigmoid) -> gate.
// 8 waves, each wave owns a 16-pixel strip (conv chain is per-pixel).
// LDS u32 rows (two f16 channels packed) padded to stride 130 dwords.
// ---------------------------------------------------------------------------
#define EPSTR 130

static __device__ __forceinline__ unsigned pack2h(float a, float b) {
  _Float16 ha = (_Float16)a, hb = (_Float16)b;
  unsigned short ua = __builtin_bit_cast(unsigned short, ha);
  unsigned short ub = __builtin_bit_cast(unsigned short, hb);
  return (unsigned)ua | ((unsigned)ub << 16);
}

__global__ __launch_bounds__(256, 1) void fused_kernel(
    const float* __restrict__ x, const _Float16* __restrict__ edge_t,
    const _Float16* __restrict__ W1f, const float* __restrict__ bias1f,
    const _Float16* __restrict__ W2f, const float* __restrict__ bias2f,
    float* __restrict__ out) {
  extern __shared__ unsigned sm2[];
  unsigned* ep = sm2;                    // [128 cpair][130]  edge, pair-interleaved
  unsigned* hp = sm2 + 128 * EPSTR;      // [ 64 cpair][130]  hidden activations

  const int bh = blockIdx.x;             // b*128 + h
  const int b  = bh >> 7;
  const int h  = bh & 127;
  const _Float16* erow = edge_t + (size_t)bh * (CIN * WW);   // [c][w], contiguous 64KB

  // ---- repack edge row: ep[c/2][w] = {edge[2c'][w], edge[2c'+1][w]} ----
  for (int i = threadIdx.x; i < 128 * 64; i += 256) {
    int cp = i >> 6, wp = i & 63;                       // wp = pixel pair
    unsigned a  = *(const unsigned*)(erow + (2 * cp)     * WW + 2 * wp);
    unsigned bq = *(const unsigned*)(erow + (2 * cp + 1) * WW + 2 * wp);
    ep[cp * EPSTR + 2 * wp]     = (a & 0xFFFFu) | (bq << 16);
    ep[cp * EPSTR + 2 * wp + 1] = (a >> 16)     | (bq & 0xFFFF0000u);
  }
  __syncthreads();

  const int lane = threadIdx.x & 31;
  const int wv   = threadIdx.x >> 5;     // 8 waves
  const int hi16 = lane >> 4;            // lane half (ISA B/C/D row +8, A K-swap)
  const int pxg  = wv * 16 + (lane & 15);

  // ================= GEMM1: h[128][px] = relu(W1f * edge + b1f) =============
  v8f acc1[8];
#pragma unroll
  for (int mt = 0; mt < 8; ++mt) acc1[mt] = (v8f){0, 0, 0, 0, 0, 0, 0, 0};

#pragma unroll
  for (int ks = 0; ks < 8; ++ks) {
    const int k0 = ks * 32;
    union { v16h v; unsigned u[8]; } bf;               // B: 32(K) x 16(px)
    const int krow = (k0 >> 1) + hi16 * 8;
#pragma unroll
    for (int j = 0; j < 8; ++j) bf.u[j] = ep[(krow + j) * EPSTR + pxg];
#pragma unroll
    for (int mt = 0; mt < 8; ++mt) {
      union { v16h v; u32x4 u[2]; } af;                // A: 16(M) x 32(K)
      const int m = mt * 16 + (lane & 15);
      const _Float16* wr = W1f + m * CIN + k0 + hi16 * 16;
      af.u[0] = *(const u32x4*)(wr);
      af.u[1] = *(const u32x4*)(wr + 8);
      acc1[mt] = __builtin_amdgcn_wmma_f32_16x16x32_f16(
          false, af.v, false, bf.v, (short)0, acc1[mt], false, false);
    }
  }

  // epilogue 1: bias + relu -> f16 pairs into hp (wave-local columns)
#pragma unroll
  for (int mt = 0; mt < 8; ++mt) {
    const int mbase = mt * 16 + hi16 * 8;
#pragma unroll
    for (int r = 0; r < 8; r += 2) {
      float t0 = fmaxf(acc1[mt][r]     + bias1f[mbase + r],     0.f);
      float t1 = fmaxf(acc1[mt][r + 1] + bias1f[mbase + r + 1], 0.f);
      hp[((mbase + r) >> 1) * EPSTR + pxg] = pack2h(t0, t1);
    }
  }
  __syncthreads();

  // ================= GEMM2: w[256][px] = sigmoid(W2f * h + b2f) =============
  v8f acc2[16];
#pragma unroll
  for (int mt = 0; mt < 16; ++mt) acc2[mt] = (v8f){0, 0, 0, 0, 0, 0, 0, 0};

#pragma unroll
  for (int ks = 0; ks < 4; ++ks) {
    const int k0 = ks * 32;
    union { v16h v; unsigned u[8]; } bf;
    const int krow = (k0 >> 1) + hi16 * 8;
#pragma unroll
    for (int j = 0; j < 8; ++j) bf.u[j] = hp[(krow + j) * EPSTR + pxg];
#pragma unroll
    for (int mt = 0; mt < 16; ++mt) {
      union { v16h v; u32x4 u[2]; } af;
      const int m = mt * 16 + (lane & 15);
      const _Float16* wr = W2f + m * CH + k0 + hi16 * 16;
      af.u[0] = *(const u32x4*)(wr);
      af.u[1] = *(const u32x4*)(wr + 8);
      acc2[mt] = __builtin_amdgcn_wmma_f32_16x16x32_f16(
          false, af.v, false, bf.v, (short)0, acc2[mt], false, false);
    }
  }

  // epilogue 2: bias + sigmoid, gated residual with fresh x read
#pragma unroll
  for (int mt = 0; mt < 16; ++mt) {
#pragma unroll
    for (int r = 0; r < 8; ++r) {
      const int ch = mt * 16 + hi16 * 8 + r;
      float t = acc2[mt][r] + bias2f[ch];
      float s = 1.0f / (1.0f + __expf(-t));
      size_t gi = (((size_t)b * CIN + ch) * HH + h) * WW + pxg;
      float xv = x[gi];
      out[gi] = fmaf(s, xv, xv);
    }
  }
}

// ---------------------------------------------------------------------------
extern "C" void kernel_launch(void* const* d_in, const int* in_sizes, int n_in,
                              void* d_out, int out_size, void* d_ws, size_t ws_size,
                              hipStream_t stream) {
  (void)in_sizes; (void)n_in; (void)out_size; (void)ws_size;
  const float* x   = (const float*)d_in[0];
  const float* w1  = (const float*)d_in[1];
  const float* b1  = (const float*)d_in[2];
  const float* g1  = (const float*)d_in[3];
  const float* bt1 = (const float*)d_in[4];
  const float* m1  = (const float*)d_in[5];
  const float* v1  = (const float*)d_in[6];
  const float* w2  = (const float*)d_in[7];
  const float* b2  = (const float*)d_in[8];
  const float* g2  = (const float*)d_in[9];
  const float* bt2 = (const float*)d_in[10];
  const float* m2  = (const float*)d_in[11];
  const float* v2  = (const float*)d_in[12];
  float* out = (float*)d_out;

  char* ws = (char*)d_ws;
  _Float16* edge_t = (_Float16*)(ws + WS_EDGE);
  _Float16* W1f    = (_Float16*)(ws + WS_W1F);
  _Float16* W2f    = (_Float16*)(ws + WS_W2F);
  float*    bias1f = (float*)(ws + WS_B1F);
  float*    bias2f = (float*)(ws + WS_B2F);

  // Pass 0: BN-fold + f16 convert (65536 threads)
  fold_kernel<<<256, 256, 0, stream>>>(w1, b1, g1, bt1, m1, v1,
                                       w2, b2, g2, bt2, m2, v2,
                                       W1f, W2f, bias1f, bias2f);

  // Pass 1: edge planes (one WG per (b,c)); LDS = 2*128*129*4 B
  edge_kernel<<<BATCH * CIN, 256, 2 * HH * P1S * sizeof(float), stream>>>(x, edge_t);

  // Pass 2: fused WMMA rows (one WG per (b,h)); LDS = (128+64)*130*4 B
  fused_kernel<<<BATCH * HH, 256, (128 + 64) * EPSTR * sizeof(unsigned), stream>>>(
      x, edge_t, W1f, bias1f, W2f, bias2f, out);
}
